// Simplified_transformer_77816217469304
// MI455X (gfx1250) — compile-verified, matrix operand order checked by reading, not run on hardware
//
#include <hip/hip_runtime.h>
#include <hip/hip_bf16.h>

// CDNA5 / gfx1250: wave32, WMMA 16x16x32 f16 -> f32, TDM global->LDS
typedef __attribute__((ext_vector_type(16))) _Float16 v16h;
typedef __attribute__((ext_vector_type(4)))  _Float16 v4h;
typedef __attribute__((ext_vector_type(8)))  float    v8f;
typedef __attribute__((ext_vector_type(4)))  unsigned int v4u;
typedef __attribute__((ext_vector_type(8)))  int      v8i;
typedef __attribute__((ext_vector_type(4)))  int      v4i;

#define BB 8
#define HH 8
#define NN 8192
#define CC 512
#define MM 1024
#define DD 64
#define QK_SCALE 0.125f   // D^-0.5 = 1/8

// ---------------------------------------------------------------------------
// Tensor Data Mover: 2D f16 tile (tile == tensor window, fully in-bounds)
// global (row-major, row_stride elems) -> LDS (packed). D# per ISA 8.3/8.4.
// ---------------------------------------------------------------------------
__device__ __forceinline__ void tdm_load_2d_f16(unsigned lds_off, const _Float16* gptr,
                                                unsigned tile_x, unsigned tile_y,
                                                unsigned row_stride) {
  unsigned long long ga = (unsigned long long)(size_t)gptr;
  v4u g0;
  g0[0] = 1u;                                   // count=1, user mode, no gather
  g0[1] = lds_off;                              // lds_addr (bytes, group-relative)
  g0[2] = (unsigned)(ga & 0xFFFFFFFFu);         // global_addr[31:0]
  g0[3] = (unsigned)((ga >> 32) & 0x1FFFFFFu) | (2u << 30);   // addr[56:32] | type=2
  v8i g1;
  g1[0] = (int)(1u << 16);                                    // data_size=1 (2 bytes)
  g1[1] = (int)((tile_x & 0xFFFFu) << 16);                    // tensor_dim0[15:0]
  g1[2] = (int)(((tile_x >> 16) & 0xFFFFu) | ((tile_y & 0xFFFFu) << 16)); // td0 hi | td1 lo
  g1[3] = (int)(((tile_y >> 16) & 0xFFFFu) | ((tile_x & 0xFFFFu) << 16)); // td1 hi | tile_dim0
  g1[4] = (int)(tile_y & 0xFFFFu);                            // tile_dim1 (tile_dim2=0)
  g1[5] = (int)row_stride;                                    // tensor_dim0_stride[31:0]
  g1[6] = 0;                                                  // stride hi | dim1_stride lo
  g1[7] = 0;
  v4i gz = {0, 0, 0, 0};
#if __clang_major__ >= 23
  v8i gz8 = {0, 0, 0, 0, 0, 0, 0, 0};
  __builtin_amdgcn_tensor_load_to_lds(g0, g1, gz, gz, gz8, 0);
#else
  __builtin_amdgcn_tensor_load_to_lds(g0, g1, gz, gz, 0);
#endif
}

// ---------------------------------------------------------------------------
// Convert qk_w (f32, [2C x C] row-major, K-contiguous) to f16 once.
// ---------------------------------------------------------------------------
__global__ __launch_bounds__(256) void st77_convert_w(const float* __restrict__ w,
                                                      _Float16* __restrict__ w16) {
  int i = blockIdx.x * 256 + threadIdx.x;   // 131072 threads, 4 elems each
  int base = i * 4;
  float4 f = *(const float4*)(w + base);
  v4h h;
  h.x = (_Float16)f.x; h.y = (_Float16)f.y; h.z = (_Float16)f.z; h.w = (_Float16)f.w;
  *(v4h*)(w16 + base) = h;
}

// ---------------------------------------------------------------------------
// Pack v = feat (= x_chunk + prev_chunk_out) into per-head f16 layout
// ---------------------------------------------------------------------------
template <int HAS_PREV>
__global__ __launch_bounds__(256) void st77_pack_v(const float* __restrict__ x,
                                                   const float* __restrict__ prev,
                                                   int chunk, int prev_chunk,
                                                   _Float16* __restrict__ vg) {
  size_t idx4 = ((size_t)blockIdx.x * 256 + threadIdx.x) * 4;   // 4,194,304 elems total
  int c = (int)(idx4 & (CC - 1));
  int m = (int)((idx4 >> 9) & (MM - 1));
  int b = (int)(idx4 >> 19);
  size_t gx = ((size_t)b * NN + (size_t)chunk * MM + m) * CC + c;
  float4 f = *(const float4*)(x + gx);
  if (HAS_PREV) {
    size_t gp = ((size_t)b * NN + (size_t)prev_chunk * MM + m) * CC + c;
    float4 p = *(const float4*)(prev + gp);
    f.x += p.x; f.y += p.y; f.z += p.z; f.w += p.w;
  }
  int h = c >> 6, d = c & 63;
  v4h o;
  o.x = (_Float16)f.x; o.y = (_Float16)f.y; o.z = (_Float16)f.z; o.w = (_Float16)f.w;
  *(v4h*)(vg + ((size_t)((b * HH + h) * MM + m)) * DD + d) = o;
}

// ---------------------------------------------------------------------------
// QK projection: qk = (x_chunk + prev) @ W^T + bias  ([8192 x 512]x[512 x 1024])
// Block tile 128x64, 8 waves of 32x32 (2x2 WMMA tiles), K step 32 (f16).
// Weight tile staged via TDM (tensor_load_to_lds + s_wait_tensorcnt).
// ---------------------------------------------------------------------------
template <int HAS_PREV>
__global__ __launch_bounds__(256) void st77_qkproj(const float* __restrict__ x,
                                                   const float* __restrict__ prev,
                                                   int chunk, int prev_chunk,
                                                   const _Float16* __restrict__ w16,
                                                   const float* __restrict__ bias,
                                                   _Float16* __restrict__ qg,
                                                   _Float16* __restrict__ kg) {
  __shared__ __align__(32) _Float16 As[128 * 32];   // [row][k]
  __shared__ __align__(32) _Float16 Bs[64 * 32];    // [col][k]  (TDM destination)

  const int tid  = threadIdx.x;
  const int wid  = tid >> 5;
  const int lane = tid & 31;
  const int lh   = lane >> 4, ln = lane & 15;
  const int rowbase = blockIdx.x * 128;   // GEMM row (b*M + m)
  const int colbase = blockIdx.y * 64;    // GEMM col (0..1023)
  const int wr = (wid >> 1) * 32;         // wave row offset in tile
  const int wc = (wid & 1) * 32;          // wave col offset in tile
  const unsigned bs_lds = (unsigned)(size_t)(&Bs[0]);

  v8f acc[2][2] = {};

  for (int k0 = 0; k0 < CC; k0 += 32) {
    __syncthreads();
    // ---- TDM: stage 64x32 f16 weight tile into LDS (one DMA, wave 0 only)
    if (wid == 0) {
      tdm_load_2d_f16(bs_lds, w16 + (size_t)colbase * CC + k0, 32u, 64u, (unsigned)CC);
    }
    // ---- stage A tile (feat, converted to f16): 128 rows x 32 K
    #pragma unroll
    for (int u = 0; u < 4; ++u) {
      int f = tid + u * 256;               // 1024 float4s
      int r = f >> 3, kq = (f & 7) * 4;
      int nrow = rowbase + r;
      int b = nrow >> 10, m = nrow & 1023;
      size_t gx = ((size_t)b * NN + (size_t)chunk * MM + m) * CC + k0 + kq;
      float4 xv = *(const float4*)(x + gx);
      if (HAS_PREV) {
        size_t gp = ((size_t)b * NN + (size_t)prev_chunk * MM + m) * CC + k0 + kq;
        float4 pv = *(const float4*)(prev + gp);
        xv.x += pv.x; xv.y += pv.y; xv.z += pv.z; xv.w += pv.w;
      }
      v4h h;
      h.x = (_Float16)xv.x; h.y = (_Float16)xv.y; h.z = (_Float16)xv.z; h.w = (_Float16)xv.w;
      *(v4h*)(&As[r * 32 + kq]) = h;
      if (k0 + 32 < CC) __builtin_prefetch(x + gx + 32, 0, 1);   // global_prefetch_b8
    }
    if (wid == 0) {
      __builtin_amdgcn_s_wait_tensorcnt(0);   // TDM complete before publish
    }
    __syncthreads();

    // ---- WMMA: A frag = row (ln), K-half (lh); B frag = col (ln), K-half (lh)
    v16h a0 = *(const v16h*)(&As[(wr + ln) * 32 + lh * 16]);
    v16h a1 = *(const v16h*)(&As[(wr + 16 + ln) * 32 + lh * 16]);
    v16h b0 = *(const v16h*)(&Bs[(wc + ln) * 32 + lh * 16]);
    v16h b1 = *(const v16h*)(&Bs[(wc + 16 + ln) * 32 + lh * 16]);
    acc[0][0] = __builtin_amdgcn_wmma_f32_16x16x32_f16(false, a0, false, b0, (short)0, acc[0][0], false, false);
    acc[0][1] = __builtin_amdgcn_wmma_f32_16x16x32_f16(false, a0, false, b1, (short)0, acc[0][1], false, false);
    acc[1][0] = __builtin_amdgcn_wmma_f32_16x16x32_f16(false, a1, false, b0, (short)0, acc[1][0], false, false);
    acc[1][1] = __builtin_amdgcn_wmma_f32_16x16x32_f16(false, a1, false, b1, (short)0, acc[1][1], false, false);
  }

  // ---- epilogue: bias + scatter into q (scaled) / k per-head buffers
  #pragma unroll
  for (int ii = 0; ii < 2; ++ii) {
    #pragma unroll
    for (int jj = 0; jj < 2; ++jj) {
      #pragma unroll
      for (int r = 0; r < 8; ++r) {
        int nrow = rowbase + wr + ii * 16 + lh * 8 + r;   // C-frag: M = 8*(lane>>4)+vgpr
        int jg   = colbase + wc + jj * 16 + ln;            //          N = lane&15
        float val = acc[ii][jj][r] + bias[jg];
        int b = nrow >> 10, m = nrow & 1023;
        if (jg < CC) {
          int h = jg >> 6, d = jg & 63;
          qg[((size_t)((b * HH + h) * MM + m)) * DD + d] = (_Float16)(val * QK_SCALE);
        } else {
          int j2 = jg - CC;
          int h = j2 >> 6, d = j2 & 63;
          kg[((size_t)((b * HH + h) * MM + m)) * DD + d] = (_Float16)val;
        }
      }
    }
  }
}

// ---------------------------------------------------------------------------
// Flash attention per (b,h): each block = 128 query rows (8 waves x 16),
// loop over key blocks of 32.
// ---------------------------------------------------------------------------
__global__ __launch_bounds__(256) void st77_attn(const _Float16* __restrict__ qg,
                                                 const _Float16* __restrict__ kg,
                                                 const _Float16* __restrict__ vg,
                                                 float* __restrict__ out, int chunk) {
  __shared__ __align__(32) _Float16 vT[DD * 32];        // [d][key]   (shared)
  __shared__ __align__(32) _Float16 Pbuf[8][16 * 32];   // [row][key] (per wave)

  const int tid  = threadIdx.x;
  const int wid  = tid >> 5;
  const int lane = tid & 31;
  const int lh   = lane >> 4, ln = lane & 15;
  const int bh   = blockIdx.y;                 // b*H + h
  const int qbase = blockIdx.x * 128 + wid * 16;

  const _Float16* qh = qg + (size_t)bh * MM * DD;
  const _Float16* kh = kg + (size_t)bh * MM * DD;
  const _Float16* vh = vg + (size_t)bh * MM * DD;

  // q fragments (16 rows x 64 d) resident in VGPRs for the whole kernel
  v16h qa0 = *(const v16h*)(qh + (size_t)(qbase + ln) * DD + lh * 16);
  v16h qa1 = *(const v16h*)(qh + (size_t)(qbase + ln) * DD + 32 + lh * 16);

  float mrow[8], lrow[8];
  v8f o[4] = {};
  #pragma unroll
  for (int r = 0; r < 8; ++r) { mrow[r] = -1e30f; lrow[r] = 0.0f; }

  for (int kb = 0; kb < MM; kb += 32) {
    __syncthreads();                           // protect vT reuse
    // stage V transposed: vT[d][key]
    {
      int key = tid >> 3;                      // 0..31
      int d0  = (tid & 7) * 8;
      const _Float16* src = vh + (size_t)(kb + key) * DD + d0;
      #pragma unroll
      for (int e = 0; e < 8; ++e) vT[(d0 + e) * 32 + key] = src[e];
    }
    __syncthreads();

    // S = q @ k^T for 32 keys (two 16-key halves, K=64 -> 2 WMMAs each)
    v8f s[2];
    #pragma unroll
    for (int h2 = 0; h2 < 2; ++h2) {
      const _Float16* kr = kh + (size_t)(kb + h2 * 16 + ln) * DD;   // B: col = key
      v16h kb0 = *(const v16h*)(kr + lh * 16);
      v16h kb1 = *(const v16h*)(kr + 32 + lh * 16);
      v8f t = {};
      t = __builtin_amdgcn_wmma_f32_16x16x32_f16(false, qa0, false, kb0, (short)0, t, false, false);
      t = __builtin_amdgcn_wmma_f32_16x16x32_f16(false, qa1, false, kb1, (short)0, t, false, false);
      s[h2] = t;
    }

    // online softmax: row = 8*(lane>>4) + r, its 16 values live across a 16-lane group
    _Float16* Pw = Pbuf[wid];
    #pragma unroll
    for (int r = 0; r < 8; ++r) {
      float vmax = fmaxf(s[0][r], s[1][r]);
      #pragma unroll
      for (int mk = 8; mk >= 1; mk >>= 1) vmax = fmaxf(vmax, __shfl_xor(vmax, mk, 32));
      float mnew  = fmaxf(mrow[r], vmax);
      float alpha = __expf(mrow[r] - mnew);
      float p0 = __expf(s[0][r] - mnew);
      float p1 = __expf(s[1][r] - mnew);
      float rs = p0 + p1;
      #pragma unroll
      for (int mk = 8; mk >= 1; mk >>= 1) rs += __shfl_xor(rs, mk, 32);
      lrow[r] = lrow[r] * alpha + rs;
      mrow[r] = mnew;
      int m = lh * 8 + r;
      Pw[m * 32 + ln]      = (_Float16)p0;
      Pw[m * 32 + 16 + ln] = (_Float16)p1;
      #pragma unroll
      for (int t2 = 0; t2 < 4; ++t2) o[t2][r] *= alpha;
    }

    // O += P(16x32) @ V(32x64): A frag from wave-private LDS (in-order DS),
    // B frags from transposed V in LDS.
    v16h pa = *(const v16h*)(&Pw[ln * 32 + lh * 16]);
    #pragma unroll
    for (int t2 = 0; t2 < 4; ++t2) {
      v16h vb = *(const v16h*)(&vT[(t2 * 16 + ln) * 32 + lh * 16]);
      o[t2] = __builtin_amdgcn_wmma_f32_16x16x32_f16(false, pa, false, vb, (short)0, o[t2], false, false);
    }
  }

  // normalize and write f32 output chunk (also next iteration's carry)
  const int b = bh >> 3, h = bh & 7;
  #pragma unroll
  for (int r = 0; r < 8; ++r) {
    float inv = 1.0f / lrow[r];
    int m = qbase + lh * 8 + r;
    float* orow = out + ((size_t)b * NN + (size_t)chunk * MM + m) * CC + h * DD;
    #pragma unroll
    for (int t2 = 0; t2 < 4; ++t2) orow[t2 * 16 + ln] = o[t2][r] * inv;
  }
}

// ---------------------------------------------------------------------------
extern "C" void kernel_launch(void* const* d_in, const int* in_sizes, int n_in,
                              void* d_out, int out_size, void* d_ws, size_t ws_size,
                              hipStream_t stream) {
  const float* x    = (const float*)d_in[0];   // [B, N, C]
  const float* qk_w = (const float*)d_in[1];   // [2C, C]
  const float* qk_b = (const float*)d_in[2];   // [2C]
  float* out = (float*)d_out;                  // [B, N, C]

  char* ws = (char*)d_ws;
  _Float16* w16 = (_Float16*)(ws);                          // 1 MB  (2C*C f16)
  _Float16* qb  = (_Float16*)(ws + (size_t)1  * (1 << 20)); // 8 MB  (B*H*M*D f16)
  _Float16* kb  = (_Float16*)(ws + (size_t)9  * (1 << 20)); // 8 MB
  _Float16* vb  = (_Float16*)(ws + (size_t)17 * (1 << 20)); // 8 MB  -> 25 MB total
  (void)in_sizes; (void)n_in; (void)out_size; (void)ws_size;

  st77_convert_w<<<512, 256, 0, stream>>>(qk_w, w16);

  const dim3 gproj(NN * BB / (8 * 128), (2 * CC) / 64);
  const int gpackv = (BB * MM * CC) / (256 * 4);
  const dim3 gattn(MM / 128, BB * HH);

  for (int i = 0; i < HH; ++i) {
    const float* prev = out;   // previous chunk of output is the carry
    int pc = (i > 0) ? (i - 1) : 0;
    if (i == 0) {
      st77_qkproj<0><<<gproj, 256, 0, stream>>>(x, prev, i, pc, w16, qk_b, qb, kb);
      st77_pack_v<0><<<gpackv, 256, 0, stream>>>(x, prev, i, pc, vb);
    } else {
      st77_qkproj<1><<<gproj, 256, 0, stream>>>(x, prev, i, pc, w16, qk_b, qb, kb);
      st77_pack_v<1><<<gpackv, 256, 0, stream>>>(x, prev, i, pc, vb);
    }
    st77_attn<<<gattn, 256, 0, stream>>>(qb, kb, vb, out, i);
  }
}
